// Transformer_54176717471969
// MI455X (gfx1250) — compile-verified, matrix operand order checked by reading, not run on hardware
//
#include <hip/hip_runtime.h>
#include <hip/hip_bf16.h>
#include <cstdint>
#include <cstddef>

// Problem constants (from reference): B,S,V,D,H,HD,M,L = 2,1024,32000,1024,16,64,1024,2
#define BB   2
#define SS   1024
#define VV   32000
#define DD   1024
#define HH   16
#define HDIM 64
#define MMID 1024

typedef __attribute__((ext_vector_type(16))) __bf16 v16bf;
typedef __attribute__((ext_vector_type(8)))  float  v8f;
typedef __attribute__((ext_vector_type(4)))  unsigned u32x4;
typedef __attribute__((ext_vector_type(8)))  unsigned u32x8;

__device__ __forceinline__ __bf16 tobf(float x) { return (__bf16)x; }

__device__ __forceinline__ v8f vzero8() {
  v8f z = {0.f,0.f,0.f,0.f,0.f,0.f,0.f,0.f};
  return z;
}

// Low 32 bits of a generic pointer to a __shared__ object = LDS byte offset
// (LDS aperture base lives in the high half of the flat address).
__device__ __forceinline__ unsigned lds_off_of(const void* p) {
  return (unsigned)(uintptr_t)p;
}

// Load a 16-element bf16 fragment as two 16-byte chunks (matches the CDNA5
// 16-bit A/B per-lane packing: elems 0..7 at k=hi*8, elems 8..15 at k=16+hi*8).
template <typename P>
__device__ __forceinline__ v16bf ldfrag(P p0, P p1) {
  union { v16bf v; uint4 q[2]; } u;
  u.q[0] = *(const uint4*)p0;
  u.q[1] = *(const uint4*)p1;
  return u.v;
}

__device__ __forceinline__ v8f wmma_bf16(v16bf a, v16bf b, v8f c) {
  // (neg_a, A, neg_b, B, c_mod, C, reuse_a, reuse_b)
  return __builtin_amdgcn_wmma_f32_16x16x32_bf16(false, a, false, b, (short)0, c, false, false);
}

// Tensor Data Mover: DMA a 2D tile rows x cols of bf16 (row-major, element
// stride `srcStride`) from global memory into LDS at `ldsAddr`.  Issued once
// by the executing wave; completion tracked with TENSORcnt.
__device__ __forceinline__ void tdm_load_tile_bf16(unsigned ldsAddr, const __bf16* gsrc,
                                                   unsigned rows, unsigned cols,
                                                   unsigned srcStride) {
  unsigned long long ga = (unsigned long long)(uintptr_t)gsrc;
  u32x4 g0;
  g0[0] = 1u;                                            // count=1 (valid D#)
  g0[1] = ldsAddr;                                       // lds_addr
  g0[2] = (unsigned)(ga & 0xffffffffu);                  // global_addr[31:0]
  g0[3] = (unsigned)((ga >> 32) & 0x01ffffffu)           // global_addr[56:32]
          | (2u << 30);                                  // type=2 ("image")
  u32x8 g1;
  g1[0] = (1u << 16);        // workgroup_mask=0, data_size=1 (2 bytes)
  g1[1] = (cols << 16);      // tensor_dim0[15:0]
  g1[2] = (rows << 16);      // tensor_dim0[31:16]=0, tensor_dim1[15:0]
  g1[3] = (cols << 16);      // tensor_dim1[31:16]=0, tile_dim0
  g1[4] = rows;              // tile_dim1, tile_dim2=0
  g1[5] = srcStride;         // tensor_dim0_stride[31:0]
  g1[6] = 0u;                // stride hi, tensor_dim1_stride lo
  g1[7] = 0u;
  asm volatile("tensor_load_to_lds %0, %1" :: "s"(g0), "s"(g1) : "memory");
}

// ---------------------------------------------------------------------------
// f32 -> bf16 conversion (weights)
// ---------------------------------------------------------------------------
__global__ void cvt_f32_bf16(const float* __restrict__ src, __bf16* __restrict__ dst, size_t n) {
  size_t i = (size_t)blockIdx.x * blockDim.x + threadIdx.x;
  size_t stride = (size_t)gridDim.x * blockDim.x;
  for (; i < n; i += stride) dst[i] = tobf(src[i]);
}

// ---------------------------------------------------------------------------
// Embedding gather + sinusoidal positional encoding + LayerNorm1.
// ---------------------------------------------------------------------------
__global__ void embed_ln_kernel(const int* __restrict__ x, const float* __restrict__ tab,
                                const float* __restrict__ g, const float* __restrict__ bt,
                                float* __restrict__ emb, __bf16* __restrict__ lnb,
                                int* __restrict__ pad) {
  int tokIdx = blockIdx.x;        // b*S + s
  int s = tokIdx & (SS - 1);
  int tid = threadIdx.x;
  int tok = x[tokIdx];
  float vals[4];
  float sum = 0.f, sq = 0.f;
  #pragma unroll
  for (int i = 0; i < 4; ++i) {
    int d = tid + 256 * i;
    float e = tab[(size_t)tok * DD + d];
    int j = d >> 1;
    float ang = (float)s * __expf(-(float)(2 * j) * (9.210340371976184f / (float)DD));
    e += (d & 1) ? __cosf(ang) : __sinf(ang);
    vals[i] = e; sum += e; sq += e * e;
  }
  #pragma unroll
  for (int off = 16; off > 0; off >>= 1) {
    sum += __shfl_xor(sum, off, 32);
    sq  += __shfl_xor(sq,  off, 32);
  }
  __shared__ float rs[8], rq[8];
  int wid = tid >> 5, lane = tid & 31;
  if (lane == 0) { rs[wid] = sum; rq[wid] = sq; }
  __syncthreads();
  float ts = 0.f, tq = 0.f;
  #pragma unroll
  for (int i = 0; i < 8; ++i) { ts += rs[i]; tq += rq[i]; }
  float mu   = ts / (float)DD;
  float var  = tq / (float)DD - mu * mu;
  float rstd = rsqrtf(var + 1e-5f);
  #pragma unroll
  for (int i = 0; i < 4; ++i) {
    int d = tid + 256 * i;
    size_t o = (size_t)tokIdx * DD + d;
    emb[o] = vals[i];
    lnb[o] = tobf((vals[i] - mu) * rstd * g[d] + bt[d]);
  }
  if (tid == 0) pad[tokIdx] = (tok == 0) ? 1 : 0;
}

// ---------------------------------------------------------------------------
// residual add + LayerNorm2 + ReLU -> bf16 (input of MLP)
// ---------------------------------------------------------------------------
__global__ void add_ln_relu_kernel(const float* __restrict__ attn, const float* __restrict__ emb,
                                   const float* __restrict__ g, const float* __restrict__ bt,
                                   __bf16* __restrict__ hb) {
  int tokIdx = blockIdx.x;
  int tid = threadIdx.x;
  float vals[4];
  float sum = 0.f, sq = 0.f;
  #pragma unroll
  for (int i = 0; i < 4; ++i) {
    int d = tid + 256 * i;
    size_t o = (size_t)tokIdx * DD + d;
    float e = attn[o] + emb[o];
    vals[i] = e; sum += e; sq += e * e;
  }
  #pragma unroll
  for (int off = 16; off > 0; off >>= 1) {
    sum += __shfl_xor(sum, off, 32);
    sq  += __shfl_xor(sq,  off, 32);
  }
  __shared__ float rs[8], rq[8];
  int wid = tid >> 5, lane = tid & 31;
  if (lane == 0) { rs[wid] = sum; rq[wid] = sq; }
  __syncthreads();
  float ts = 0.f, tq = 0.f;
  #pragma unroll
  for (int i = 0; i < 8; ++i) { ts += rs[i]; tq += rq[i]; }
  float mu   = ts / (float)DD;
  float var  = tq / (float)DD - mu * mu;
  float rstd = rsqrtf(var + 1e-5f);
  #pragma unroll
  for (int i = 0; i < 4; ++i) {
    int d = tid + 256 * i;
    size_t o = (size_t)tokIdx * DD + d;
    float l = (vals[i] - mu) * rstd * g[d] + bt[d];
    hb[o] = tobf(fmaxf(l, 0.f));
  }
}

// ---------------------------------------------------------------------------
// Per-head QKV projection: [S,64] x [64,64] + bias, via bf16 WMMA.
// q,k stored [B,H,S,HD]; v stored transposed [B,H,HD,S] (for P·V B-fragments).
// ---------------------------------------------------------------------------
__global__ void qkv_kernel(const __bf16* __restrict__ lnb,
                           const float* __restrict__ wq, const float* __restrict__ bq,
                           const float* __restrict__ wk, const float* __restrict__ bk,
                           const float* __restrict__ wv, const float* __restrict__ bv,
                           __bf16* __restrict__ qb, __bf16* __restrict__ kb,
                           __bf16* __restrict__ vT) {
  int z  = blockIdx.z;
  int bh = blockIdx.y;
  int b = bh / HH, h = bh % HH;
  const float* wsel = (z == 0 ? wq : (z == 1 ? wk : wv)) + (size_t)h * HDIM * HDIM;
  const float* bsel = (z == 0 ? bq : (z == 1 ? bk : bv)) + h * HDIM;

  __shared__ __bf16 WT[64][64];   // WT[e][d] = w[d][e]
  __shared__ float  bs[64];
  int tid = threadIdx.x;
  #pragma unroll
  for (int i = 0; i < 16; ++i) {
    int idx = tid + 256 * i;
    int d = idx >> 6, e = idx & 63;
    WT[e][d] = tobf(wsel[idx]);
  }
  if (tid < 64) bs[tid] = bsel[tid];
  __syncthreads();

  int wid = tid >> 5, lane = tid & 31;
  int nl = lane & 15, hi = lane >> 4;
  int sbase = blockIdx.x * 128 + wid * 16;

  const __bf16* arow = lnb + ((size_t)(b * SS + sbase + nl)) * DD + h * HDIM;
  v8f acc[4] = {vzero8(), vzero8(), vzero8(), vzero8()};
  #pragma unroll
  for (int j = 0; j < 2; ++j) {
    v16bf af = ldfrag(arow + 32 * j + hi * 8, arow + 32 * j + 16 + hi * 8);
    #pragma unroll
    for (int d = 0; d < 4; ++d) {
      v16bf wf = ldfrag(&WT[16 * d + nl][32 * j + hi * 8],
                        &WT[16 * d + nl][32 * j + 16 + hi * 8]);
      acc[d] = wmma_bf16(af, wf, acc[d]);
    }
  }
  #pragma unroll
  for (int d = 0; d < 4; ++d) {
    int n = 16 * d + nl;
    float bias = bs[n];
    #pragma unroll
    for (int r = 0; r < 8; ++r) {
      int srow = sbase + hi * 8 + r;
      float val = acc[d][r] + bias;
      if (z == 0)      qb[((size_t)bh * SS + srow) * HDIM + n] = tobf(val);
      else if (z == 1) kb[((size_t)bh * SS + srow) * HDIM + n] = tobf(val);
      else             vT[((size_t)bh * HDIM + n) * SS + srow] = tobf(val);
    }
  }
}

// ---------------------------------------------------------------------------
// Flash attention with the reference's (buggy) mask: masked where t<=s OR pad[t].
// Pad flags staged global->LDS through the CDNA5 async path (ASYNCcnt).
// ---------------------------------------------------------------------------
__global__ void attn_kernel(const __bf16* __restrict__ qb, const __bf16* __restrict__ kb,
                            const __bf16* __restrict__ vT, const int* __restrict__ pad,
                            float* __restrict__ outp) {
  int bh = blockIdx.y;
  int b = bh / HH, h = bh % HH;
  int tid = threadIdx.x;
  int wid = tid >> 5, lane = tid & 31;
  int nl = lane & 15, hi = lane >> 4;
  int qbase = blockIdx.x * 128 + wid * 16;

  __shared__ __bf16 Plds[8][16][32];   // per-wave P tile for C-layout -> A-layout
  __shared__ int pads[SS];

  // async global->LDS copy: 256 threads x 16B = 4KB (whole pad vector)
  {
    unsigned ldsa = lds_off_of(&pads[tid * 4]);
    const int* gsrc = pad + b * SS + tid * 4;
    asm volatile("global_load_async_to_lds_b128 %0, %1, off"
                 :: "v"(ldsa), "v"(gsrc) : "memory");
    asm volatile("s_wait_asynccnt 0x0" ::: "memory");
  }
  __syncthreads();

  const __bf16* qrow = qb + ((size_t)bh * SS + qbase + nl) * HDIM;
  v16bf aq0 = ldfrag(qrow + hi * 8,      qrow + 16 + hi * 8);
  v16bf aq1 = ldfrag(qrow + 32 + hi * 8, qrow + 48 + hi * 8);

  v8f o[4] = {vzero8(), vzero8(), vzero8(), vzero8()};
  float mrun[8], lrun[8];
  #pragma unroll
  for (int r = 0; r < 8; ++r) { mrun[r] = -3.0e38f; lrun[r] = 0.f; }

  for (int ktb = 0; ktb < SS; ktb += 32) {
    v8f sc[2];
    int pd[2];
    #pragma unroll
    for (int c = 0; c < 2; ++c) {
      int t = ktb + 16 * c + nl;
      const __bf16* krow = kb + ((size_t)bh * SS + t) * HDIM;
      v16bf bk0 = ldfrag(krow + hi * 8,      krow + 16 + hi * 8);
      v16bf bk1 = ldfrag(krow + 32 + hi * 8, krow + 48 + hi * 8);
      v8f s0 = wmma_bf16(aq0, bk0, vzero8());
      sc[c]  = wmma_bf16(aq1, bk1, s0);
      pd[c]  = pads[t];
    }
    int t0 = ktb + nl, t1 = ktb + 16 + nl;
    #pragma unroll
    for (int r = 0; r < 8; ++r) {
      int srow = qbase + hi * 8 + r;
      float v0 = sc[0][r] * 0.125f;     // 1/sqrt(64)
      float v1 = sc[1][r] * 0.125f;
      if (t0 <= srow || pd[0]) v0 = -1e9f;
      if (t1 <= srow || pd[1]) v1 = -1e9f;
      float mx = fmaxf(v0, v1);
      #pragma unroll
      for (int off = 8; off > 0; off >>= 1) mx = fmaxf(mx, __shfl_xor(mx, off, 32));
      float mnew = fmaxf(mrun[r], mx);
      float corr = __expf(mrun[r] - mnew);
      mrun[r] = mnew;
      float p0 = __expf(v0 - mnew);
      float p1 = __expf(v1 - mnew);
      float rsum = p0 + p1;
      #pragma unroll
      for (int off = 8; off > 0; off >>= 1) rsum += __shfl_xor(rsum, off, 32);
      lrun[r] = lrun[r] * corr + rsum;
      #pragma unroll
      for (int d = 0; d < 4; ++d) o[d][r] *= corr;
      Plds[wid][hi * 8 + r][nl]      = tobf(p0);
      Plds[wid][hi * 8 + r][16 + nl] = tobf(p1);
    }
    v16bf pf = ldfrag(&Plds[wid][nl][hi * 8], &Plds[wid][nl][16 + hi * 8]);
    #pragma unroll
    for (int d = 0; d < 4; ++d) {
      const __bf16* vrow = vT + ((size_t)bh * HDIM + 16 * d + nl) * SS + ktb;
      v16bf bvf = ldfrag(vrow + hi * 8, vrow + 16 + hi * 8);
      o[d] = wmma_bf16(pf, bvf, o[d]);
    }
  }
  #pragma unroll
  for (int d = 0; d < 4; ++d) {
    int n = 16 * d + nl;
    #pragma unroll
    for (int r = 0; r < 8; ++r) {
      int srow = qbase + hi * 8 + r;
      outp[((size_t)b * SS + srow) * DD + h * HDIM + n] = o[d][r] / lrun[r];
    }
  }
}

// ---------------------------------------------------------------------------
// Generic bf16 GEMM: C[Mtot,N] = A[Mtot,K] x W[K,N] (+bias) [+relu+resid].
// Block tile 64x128, 8 waves (4x2), each wave 16x64 via 4 accumulators.
// A tile (64x32, row-major, stride K) is DMA'd into LDS by the Tensor Data
// Mover (TENSORcnt); W tile is transpose-staged by the waves in parallel
// with the DMA.  mode 0: out = AW + bias.  mode 1: out = relu(AW+bias)+resid.
// ---------------------------------------------------------------------------
__global__ void gemm_kernel(const __bf16* __restrict__ A, const __bf16* __restrict__ W,
                            const float* __restrict__ bias, const float* __restrict__ resid,
                            float* __restrict__ outF, __bf16* __restrict__ outB,
                            int N, int K, int mode) {
  __shared__ __bf16 Asl[64][32];
  __shared__ __bf16 Wsl[128][32];   // transposed: Wsl[n][k]
  int tid = threadIdx.x;
  int wid = tid >> 5, lane = tid & 31;
  int nl = lane & 15, hi = lane >> 4;
  int wm = wid >> 1, wn = wid & 1;
  int m0 = blockIdx.y * 64;
  int n0 = blockIdx.x * 128;
  v8f acc[4] = {vzero8(), vzero8(), vzero8(), vzero8()};
  unsigned aslBase = lds_off_of(&Asl[0][0]);

  for (int k0 = 0; k0 < K; k0 += 32) {
    // kick off the TDM DMA for the A tile (one issue, wave 0)
    if (wid == 0) {
      tdm_load_tile_bf16(aslBase, A + (size_t)m0 * K + k0, 64u, 32u, (unsigned)K);
    }
    // meanwhile all 8 waves transpose-stage the W tile
    #pragma unroll
    for (int i = 0; i < 2; ++i) {
      int idx = tid + 256 * i;
      int kk = idx >> 4, nq = idx & 15;
      union { uint4 q; __bf16 hh[8]; } u;
      u.q = *(const uint4*)&W[(size_t)(k0 + kk) * N + n0 + nq * 8];
      #pragma unroll
      for (int j = 0; j < 8; ++j) Wsl[nq * 8 + j][kk] = u.hh[j];
    }
    if (wid == 0) __builtin_amdgcn_s_wait_tensorcnt(0);
    __syncthreads();

    v16bf af = ldfrag(&Asl[wm * 16 + nl][hi * 8], &Asl[wm * 16 + nl][16 + hi * 8]);
    #pragma unroll
    for (int d = 0; d < 4; ++d) {
      v16bf wf = ldfrag(&Wsl[wn * 64 + 16 * d + nl][hi * 8],
                        &Wsl[wn * 64 + 16 * d + nl][16 + hi * 8]);
      acc[d] = wmma_bf16(af, wf, acc[d]);
    }
    __syncthreads();
  }

  // epilogue: uniform branches hoisted out of the store loops
  float vals[4][8];
  #pragma unroll
  for (int d = 0; d < 4; ++d) {
    int n = n0 + wn * 64 + 16 * d + nl;
    float bv = bias[n];
    #pragma unroll
    for (int r = 0; r < 8; ++r) vals[d][r] = acc[d][r] + bv;
  }
  if (mode == 1) {
    #pragma unroll
    for (int d = 0; d < 4; ++d) {
      int n = n0 + wn * 64 + 16 * d + nl;
      #pragma unroll
      for (int r = 0; r < 8; ++r) {
        int m = m0 + wm * 16 + hi * 8 + r;
        vals[d][r] = fmaxf(vals[d][r], 0.f) + resid[(size_t)m * N + n];
      }
    }
  }
  #pragma unroll
  for (int d = 0; d < 4; ++d) {
    int n = n0 + wn * 64 + 16 * d + nl;
    #pragma unroll
    for (int r = 0; r < 8; ++r) {
      int m = m0 + wm * 16 + hi * 8 + r;
      outF[(size_t)m * N + n] = vals[d][r];
    }
  }
  if (outB != nullptr) {
    #pragma unroll
    for (int d = 0; d < 4; ++d) {
      int n = n0 + wn * 64 + 16 * d + nl;
      #pragma unroll
      for (int r = 0; r < 8; ++r) {
        int m = m0 + wm * 16 + hi * 8 + r;
        outB[(size_t)m * N + n] = tobf(vals[d][r]);
      }
    }
  }
}

// ---------------------------------------------------------------------------
extern "C" void kernel_launch(void* const* d_in, const int* in_sizes, int n_in,
                              void* d_out, int out_size, void* d_ws, size_t ws_size,
                              hipStream_t stream) {
  (void)in_sizes; (void)n_in; (void)out_size; (void)ws_size;
  const int*   x      = (const int*)  d_in[0];
  const float* tab    = (const float*)d_in[1];
  const float* ln1_g  = (const float*)d_in[2];
  const float* ln1_b  = (const float*)d_in[3];
  const float* wq     = (const float*)d_in[4];
  const float* bq     = (const float*)d_in[5];
  const float* wk     = (const float*)d_in[6];
  const float* bk     = (const float*)d_in[7];
  const float* wv     = (const float*)d_in[8];
  const float* bv     = (const float*)d_in[9];
  const float* ln2_g  = (const float*)d_in[10];
  const float* ln2_b  = (const float*)d_in[11];
  const float* w_in   = (const float*)d_in[12];
  const float* b_in   = (const float*)d_in[13];
  const float* w_mid  = (const float*)d_in[14];
  const float* b_mid  = (const float*)d_in[15];
  const float* w_out  = (const float*)d_in[16];
  const float* b_out  = (const float*)d_in[17];
  float* logits = (float*)d_out;

  // bump allocator over d_ws
  char* base = (char*)d_ws;
  size_t off = 0;
  auto alloc = [&](size_t bytes) -> char* {
    char* p = base + off;
    off += (bytes + 255) & ~(size_t)255;
    return p;
  };
  const size_t NTOK = (size_t)BB * SS;
  float*  emb   = (float*) alloc(NTOK * DD * 4);
  __bf16* lnb   = (__bf16*)alloc(NTOK * DD * 2);
  int*    pad   = (int*)   alloc(NTOK * 4);
  __bf16* qb    = (__bf16*)alloc((size_t)BB * HH * SS * HDIM * 2);
  __bf16* kb    = (__bf16*)alloc((size_t)BB * HH * SS * HDIM * 2);
  __bf16* vT    = (__bf16*)alloc((size_t)BB * HH * HDIM * SS * 2);
  float*  attn  = (float*) alloc(NTOK * DD * 4);
  __bf16* hb0   = (__bf16*)alloc(NTOK * MMID * 2);
  __bf16* hb1   = (__bf16*)alloc(NTOK * MMID * 2);
  float*  hres  = (float*) alloc(NTOK * MMID * 4);
  __bf16* w_inb  = (__bf16*)alloc((size_t)DD * MMID * 2);
  __bf16* w_midb = (__bf16*)alloc((size_t)2 * MMID * MMID * 2);
  __bf16* w_outb = (__bf16*)alloc((size_t)MMID * VV * 2);

  // weight conversions (f32 -> bf16)
  cvt_f32_bf16<<<2048, 256, 0, stream>>>(w_in,  w_inb,  (size_t)DD * MMID);
  cvt_f32_bf16<<<4096, 256, 0, stream>>>(w_mid, w_midb, (size_t)2 * MMID * MMID);
  cvt_f32_bf16<<<8192, 256, 0, stream>>>(w_out, w_outb, (size_t)MMID * VV);

  // embedding + posenc + LN1
  embed_ln_kernel<<<BB * SS, 256, 0, stream>>>(x, tab, ln1_g, ln1_b, emb, lnb, pad);

  // QKV projections (WMMA)
  qkv_kernel<<<dim3(SS / 128, BB * HH, 3), 256, 0, stream>>>(
      lnb, wq, bq, wk, bk, wv, bv, qb, kb, vT);

  // flash attention (WMMA + async global->LDS)
  attn_kernel<<<dim3(SS / 128, BB * HH), 256, 0, stream>>>(qb, kb, vT, pad, attn);

  // residual + LN2 + ReLU
  add_ln_relu_kernel<<<BB * SS, 256, 0, stream>>>(attn, emb, ln2_g, ln2_b, hb0);

  // MLP: h1 = h0 @ w_in + b_in
  gemm_kernel<<<dim3(MMID / 128, (int)(NTOK / 64)), 256, 0, stream>>>(
      hb0, w_inb, b_in, nullptr, hres, hb1, MMID, DD, 0);
  // h2 = relu(h1 @ w_mid0 + b0) + h1
  gemm_kernel<<<dim3(MMID / 128, (int)(NTOK / 64)), 256, 0, stream>>>(
      hb1, w_midb, b_mid, hres, hres, hb0, MMID, MMID, 1);
  // h3 = relu(h2 @ w_mid1 + b1) + h2
  gemm_kernel<<<dim3(MMID / 128, (int)(NTOK / 64)), 256, 0, stream>>>(
      hb0, w_midb + (size_t)MMID * MMID, b_mid + MMID, hres, hres, hb1, MMID, MMID, 1);
  // logits = h3 @ w_out + b_out  (w_out bf16 = 64MB, resident in 192MB L2)
  gemm_kernel<<<dim3(VV / 128, (int)(NTOK / 64)), 256, 0, stream>>>(
      hb1, w_outb, b_out, nullptr, logits, nullptr, VV, DD, 0);
}
